// PFNN_13795434955174
// MI455X (gfx1250) — matrix operand order repaired and checked by reading, not compiled, not run on hardware
//
#include <hip/hip_runtime.h>
#include <math.h>

// Problem constants from the reference
#define B_    1024
#define DIN_  343
#define H_    512
#define DOUT_ 311
#define NC_   4

// Padded shapes (K -> multiple of 32 for BK, N -> multiple of 128 for BN)
#define K0P_  352      // 342 padded
#define N2P_  384      // 311 padded

typedef __attribute__((ext_vector_type(16))) __bf16 v16bf;
typedef __attribute__((ext_vector_type(8)))  __bf16 v8bf;
typedef __attribute__((ext_vector_type(8)))  float  v8f;

union FragAB { v16bf f; v8bf h[2]; };

// ---- CDNA5 async global->LDS copy engine (guarded; falls back to reg copy)
#if defined(__gfx1250__) && __has_builtin(__builtin_amdgcn_global_load_async_to_lds_b128)
#define PFNN_ASYNC 1
#else
#define PFNN_ASYNC 0
#endif

// exact parameter type from the builtin's prototype: v4i in addrspace(1)/(3)
typedef int pfnn_v4i __attribute__((vector_size(16)));
typedef __attribute__((address_space(1))) pfnn_v4i* pfnn_gptr;
typedef __attribute__((address_space(3))) pfnn_v4i* pfnn_lptr;

__device__ __forceinline__ void cp16_g2l(const __bf16* g, __bf16* l)
{
#if PFNN_ASYNC
    __builtin_amdgcn_global_load_async_to_lds_b128(
        (pfnn_gptr)g, (pfnn_lptr)l, 0 /*offset*/, 0 /*cpol*/);
#else
    *(v8bf*)l = *(const v8bf*)g;
#endif
}

__device__ __forceinline__ void stage_wait()
{
#if PFNN_ASYNC
#if __has_builtin(__builtin_amdgcn_s_wait_asynccnt)
    __builtin_amdgcn_s_wait_asynccnt(0);
#else
    asm volatile("s_wait_asynccnt 0" ::: "memory");
#endif
#endif
}

// ---------------------------------------------------------------------------
// Per-sample cubic weights: alpha[b][c] so that  W_interp(b) = sum_c alpha_c W[c]
// ---------------------------------------------------------------------------
__global__ void pfnn_alpha_kernel(const float* __restrict__ x,
                                  float* __restrict__ alpha)
{
    int b = blockIdx.x * blockDim.x + threadIdx.x;
    if (b >= B_) return;
    float phase = x[b * DIN_ + (DIN_ - 1)];
    float ps = (float)NC_ * phase;
    float mu = ps - floorf(ps);
    int k1 = ((int)ps) & 3;          // ps in [0,4)
    int k0 = (k1 + 3) & 3;
    int k2 = (k1 + 1) & 3;
    int k3 = (k1 + 2) & 3;
    float mu2 = mu * mu, mu3 = mu2 * mu;
    float c0 = -0.5f * mu3 +        mu2 - 0.5f * mu;
    float c1 =  1.5f * mu3 - 2.5f * mu2 + 1.0f;
    float c2 = -1.5f * mu3 + 2.0f * mu2 + 0.5f * mu;
    float c3 =  0.5f * mu3 - 0.5f * mu2;
    float a4[4];
    a4[k0] = c0; a4[k1] = c1; a4[k2] = c2; a4[k3] = c3;
    ((float4*)alpha)[b] = make_float4(a4[0], a4[1], a4[2], a4[3]);
}

// ---------------------------------------------------------------------------
// f32 -> bf16 hi/lo split with zero padding.
// src layout: [C][Nsrc] rows of srcLd floats, Kvalid valid columns.
// dst layout: [C][Ndst][Kdst] bf16 (hi and lo planes), zero beyond valid region.
// ---------------------------------------------------------------------------
__global__ void pfnn_cvt_kernel(const float* __restrict__ src,
                                __bf16* __restrict__ dh, __bf16* __restrict__ dl,
                                int Nsrc, int Ndst, int srcLd, int Kvalid,
                                int Kdst, int total)
{
    int idx = blockIdx.x * blockDim.x + threadIdx.x;
    if (idx >= total) return;
    int k = idx % Kdst;
    int t = idx / Kdst;
    int n = t % Ndst;
    int c = t / Ndst;
    float v = (n < Nsrc && k < Kvalid)
                ? src[((size_t)c * Nsrc + n) * srcLd + k] : 0.0f;
    __bf16 hi = (__bf16)v;
    __bf16 lo = (__bf16)(v - (float)hi);
    dh[idx] = hi;
    dl[idx] = lo;
}

// ---------------------------------------------------------------------------
// Fused 4-control-point GEMM layer on pre-split bf16 planes.
//   accS_c = (A @ W_c^T) in bf16 hi/lo x3 WMMA, f32 accumulate
//   accT  += alpha_c(row) * accS_c            (per c-segment, f32)
//   out    = act(accT + sum_c alpha_c * bias_c)
// Block tile BM=64 x BN=128, BK=32, 256 threads (8 wave32s, 2x4 wave grid).
// Double-buffered LDS; staging via async global->LDS b128 copies (CDNA5
// ASYNCcnt path) when available, else b128 register copies.
// ---------------------------------------------------------------------------
template <bool ELU, bool BF16OUT>
__global__ void __launch_bounds__(256)
pfnn_layer_kernel(const __bf16* __restrict__ Ah, const __bf16* __restrict__ Al,
                  int Kp,                               // A: [B][Kp]
                  const __bf16* __restrict__ Wh, const __bf16* __restrict__ Wl,
                  int Np,                               // W: [NC][Np][Kp]
                  const float* __restrict__ bias, int N,  // f32 [NC][N]
                  const float* __restrict__ alpha,        // f32 [B][4]
                  float* __restrict__ outF,
                  __bf16* __restrict__ outH, __bf16* __restrict__ outL,
                  int ldo)
{
    constexpr int BM = 64, BN = 128, LS = 40;  // LDS row stride (bf16)
    __shared__ __align__(16) __bf16 AsH[2][BM * LS];
    __shared__ __align__(16) __bf16 AsL[2][BM * LS];
    __shared__ __align__(16) __bf16 BsH[2][BN * LS];
    __shared__ __align__(16) __bf16 BsL[2][BN * LS];

    const int tid   = threadIdx.x;
    const int lane  = tid & 31;
    const int wave  = tid >> 5;
    const int wr    = wave & 1;     // wave row: 0..1  (32 rows each)
    const int wc    = wave >> 1;    // wave col: 0..3  (32 cols each)
    const int blockRow = blockIdx.y * BM;
    const int blockCol = blockIdx.x * BN;

    const int l15  = lane & 15;
    const int half = lane >> 4;     // 0 or 1

    // staging coordinates (pure 16B copies)
    const int ar = tid >> 2;            // 0..63
    const int ak = (tid & 3) * 8;       // 0,8,16,24
    const size_t aGB = (size_t)(blockRow + ar) * Kp + ak;   // + kk
    const int    aLD = ar * LS + ak;

    // stage one 64x32 A tile + 128x32 B tile (hi+lo planes) into buffer p
    auto stage = [&](int p, int kk, const __bf16* WhC, const __bf16* WlC) {
        cp16_g2l(&Ah[aGB + kk], &AsH[p][aLD]);
        cp16_g2l(&Al[aGB + kk], &AsL[p][aLD]);
        #pragma unroll
        for (int i = 0; i < 2; ++i) {
            int idx = i * 256 + tid;
            int n  = idx >> 2;
            int kc = (idx & 3) * 8;
            size_t g = (size_t)(blockCol + n) * Kp + kk + kc;
            int    l = n * LS + kc;
            cp16_g2l(&WhC[g], &BsH[p][l]);
            cp16_g2l(&WlC[g], &BsL[p][l]);
        }
    };

    v8f accT[2][2] = {};

    for (int c = 0; c < NC_; ++c) {
        const __bf16* WhC = Wh + (size_t)c * Np * Kp;
        const __bf16* WlC = Wl + (size_t)c * Np * Kp;

        v8f accS[2][2] = {};

        // prologue: fill buffer 0
        stage(0, 0, WhC, WlC);
        stage_wait();
        __syncthreads();

        for (int kk = 0; kk < Kp; kk += 32) {
            const int p = (kk >> 5) & 1;

            // issue async staging for the next k-step into the idle buffer
            if (kk + 32 < Kp)
                stage(p ^ 1, kk + 32, WhC, WlC);

            // ---- fragment loads (16B-aligned ds_load_b128 chunks)
            // A 16x32 bf16 layout: lane m=l15, elems 0..7 -> K=kb..kb+7,
            //                      elems 8..15 -> K=16+kb..16+kb+7, kb=half*8
            FragAB fah[2], fal[2], fbh[2], fbl[2];
            #pragma unroll
            for (int mt = 0; mt < 2; ++mt) {
                int r  = wr * 32 + mt * 16 + l15;
                int kb = half * 8;
                const __bf16* ph = &AsH[p][r * LS + kb];
                fah[mt].h[0] = *(const v8bf*)(ph);
                fah[mt].h[1] = *(const v8bf*)(ph + 16);
                const __bf16* pl = &AsL[p][r * LS + kb];
                fal[mt].h[0] = *(const v8bf*)(pl);
                fal[mt].h[1] = *(const v8bf*)(pl + 16);
            }
            // B 32x16 bf16 layout: lane n=l15, elems 0..7 -> K=kb..kb+7,
            //                      elems 8..15 -> K=16+kb..16+kb+7, kb=half*8
            #pragma unroll
            for (int nt = 0; nt < 2; ++nt) {
                int n  = wc * 32 + nt * 16 + l15;
                int kb = half * 8;
                const __bf16* ph = &BsH[p][n * LS + kb];
                fbh[nt].h[0] = *(const v8bf*)(ph);
                fbh[nt].h[1] = *(const v8bf*)(ph + 16);
                const __bf16* pl = &BsL[p][n * LS + kb];
                fbl[nt].h[0] = *(const v8bf*)(pl);
                fbl[nt].h[1] = *(const v8bf*)(pl + 16);
            }

            // ---- 12 WMMAs: hi*hi + hi*lo + lo*hi per 16x16 tile
            #pragma unroll
            for (int mt = 0; mt < 2; ++mt) {
                #pragma unroll
                for (int nt = 0; nt < 2; ++nt) {
                    accS[mt][nt] = __builtin_amdgcn_wmma_f32_16x16x32_bf16(
                        false, fah[mt].f, false, fbh[nt].f, (short)0, accS[mt][nt], false, false);
                    accS[mt][nt] = __builtin_amdgcn_wmma_f32_16x16x32_bf16(
                        false, fah[mt].f, false, fbl[nt].f, (short)0, accS[mt][nt], false, false);
                    accS[mt][nt] = __builtin_amdgcn_wmma_f32_16x16x32_bf16(
                        false, fal[mt].f, false, fbh[nt].f, (short)0, accS[mt][nt], false, false);
                }
            }

            // all async writes for the idle buffer done + all waves past reads
            stage_wait();
            __syncthreads();
        }

        // ---- fold alpha_c into the running total (f32)
        // C/D layout: VGPR v, lane l: n = l&15, m = v + (l>=16 ? 8 : 0)
        #pragma unroll
        for (int mt = 0; mt < 2; ++mt) {
            float av[8];
            #pragma unroll
            for (int v = 0; v < 8; ++v) {
                int row = blockRow + wr * 32 + mt * 16 + v + half * 8;
                av[v] = alpha[row * 4 + c];
            }
            #pragma unroll
            for (int nt = 0; nt < 2; ++nt)
                #pragma unroll
                for (int v = 0; v < 8; ++v)
                    accT[mt][nt][v] += av[v] * accS[mt][nt][v];
        }
    }

    // ---- epilogue: interpolated bias + ELU, store f32 or bf16 hi/lo planes
    #pragma unroll
    for (int mt = 0; mt < 2; ++mt) {
        #pragma unroll
        for (int nt = 0; nt < 2; ++nt) {
            int col = blockCol + wc * 32 + nt * 16 + l15;
            if (col < N) {
                float b0v = bias[0 * N + col];
                float b1v = bias[1 * N + col];
                float b2v = bias[2 * N + col];
                float b3v = bias[3 * N + col];
                #pragma unroll
                for (int v = 0; v < 8; ++v) {
                    int row = blockRow + wr * 32 + mt * 16 + v + half * 8;
                    const float4 a = ((const float4*)alpha)[row];
                    float val = accT[mt][nt][v]
                              + a.x * b0v + a.y * b1v + a.z * b2v + a.w * b3v;
                    if (ELU) val = (val > 0.0f) ? val : expm1f(val);
                    if (BF16OUT) {
                        __bf16 hi = (__bf16)val;
                        __bf16 lo = (__bf16)(val - (float)hi);
                        outH[(size_t)row * ldo + col] = hi;
                        outL[(size_t)row * ldo + col] = lo;
                    } else {
                        outF[(size_t)row * ldo + col] = val;
                    }
                }
            }
        }
    }
}

// ---------------------------------------------------------------------------
extern "C" void kernel_launch(void* const* d_in, const int* in_sizes, int n_in,
                              void* d_out, int out_size, void* d_ws, size_t ws_size,
                              hipStream_t stream)
{
    const float* x  = (const float*)d_in[0];
    const float* W0 = (const float*)d_in[1];
    const float* W1 = (const float*)d_in[2];
    const float* W2 = (const float*)d_in[3];
    const float* b0 = (const float*)d_in[4];
    const float* b1 = (const float*)d_in[5];
    const float* b2 = (const float*)d_in[6];
    float* out = (float*)d_out;

    // ---- workspace carve-up (all chunks 16B-aligned by construction)
    char* w = (char*)d_ws;
    float* alpha = (float*)w;                 w += (size_t)B_ * 4 * sizeof(float);
    __bf16* Xh  = (__bf16*)w;                 w += (size_t)B_ * K0P_ * 2;
    __bf16* Xl  = (__bf16*)w;                 w += (size_t)B_ * K0P_ * 2;
    __bf16* W0h = (__bf16*)w;                 w += (size_t)NC_ * H_ * K0P_ * 2;
    __bf16* W0l = (__bf16*)w;                 w += (size_t)NC_ * H_ * K0P_ * 2;
    __bf16* W1h = (__bf16*)w;                 w += (size_t)NC_ * H_ * H_ * 2;
    __bf16* W1l = (__bf16*)w;                 w += (size_t)NC_ * H_ * H_ * 2;
    __bf16* W2h = (__bf16*)w;                 w += (size_t)NC_ * N2P_ * H_ * 2;
    __bf16* W2l = (__bf16*)w;                 w += (size_t)NC_ * N2P_ * H_ * 2;
    __bf16* H1h = (__bf16*)w;                 w += (size_t)B_ * H_ * 2;
    __bf16* H1l = (__bf16*)w;                 w += (size_t)B_ * H_ * 2;
    __bf16* H2h = (__bf16*)w;                 w += (size_t)B_ * H_ * 2;
    __bf16* H2l = (__bf16*)w;                 w += (size_t)B_ * H_ * 2;

    dim3 blk(256);

    // ---- per-sample cubic coefficients
    pfnn_alpha_kernel<<<B_ / 256, 256, 0, stream>>>(x, alpha);

    // ---- f32 -> bf16 hi/lo pre-splits (zero-padded)
    {
        int t = B_ * K0P_;                       // x: rows=B, stride DIN_, 342 valid
        pfnn_cvt_kernel<<<(t + 255) / 256, blk, 0, stream>>>(
            x, Xh, Xl, B_, B_, DIN_, DIN_ - 1, K0P_, t);
    }
    {
        int t = NC_ * H_ * K0P_;                 // W0: [4][512][342] -> [4][512][352]
        pfnn_cvt_kernel<<<(t + 255) / 256, blk, 0, stream>>>(
            W0, W0h, W0l, H_, H_, DIN_ - 1, DIN_ - 1, K0P_, t);
    }
    {
        int t = NC_ * H_ * H_;                   // W1: [4][512][512] pass-through
        pfnn_cvt_kernel<<<(t + 255) / 256, blk, 0, stream>>>(
            W1, W1h, W1l, H_, H_, H_, H_, H_, t);
    }
    {
        int t = NC_ * N2P_ * H_;                 // W2: [4][311][512] -> [4][384][512]
        pfnn_cvt_kernel<<<(t + 255) / 256, blk, 0, stream>>>(
            W2, W2h, W2l, DOUT_, N2P_, H_, H_, H_, t);
    }

    // ---- layer 0: 342 -> 512, ELU, bf16-plane output
    pfnn_layer_kernel<true, true><<<dim3(H_ / 128, B_ / 64), blk, 0, stream>>>(
        Xh, Xl, K0P_, W0h, W0l, H_, b0, H_, alpha,
        nullptr, H1h, H1l, H_);
    // ---- layer 1: 512 -> 512, ELU, bf16-plane output
    pfnn_layer_kernel<true, true><<<dim3(H_ / 128, B_ / 64), blk, 0, stream>>>(
        H1h, H1l, H_, W1h, W1l, H_, b1, H_, alpha,
        nullptr, H2h, H2l, H_);
    // ---- layer 2: 512 -> 311, linear, f32 output
    pfnn_layer_kernel<false, false><<<dim3(N2P_ / 128, B_ / 64), blk, 0, stream>>>(
        H2h, H2l, H_, W2h, W2l, N2P_, b2, DOUT_, alpha,
        out, nullptr, nullptr, DOUT_);
}
